// KnowledgeCircuit_35356170781340
// MI455X (gfx1250) — matrix operand order repaired and checked by reading, not compiled, not run on hardware
//
#include <hip/hip_runtime.h>

typedef __attribute__((ext_vector_type(16))) _Float16 v16h;
typedef __attribute__((ext_vector_type(8)))  _Float16 v8h;
typedef __attribute__((ext_vector_type(4)))  _Float16 v4h;
typedef __attribute__((ext_vector_type(8)))  float    v8f;

#define D_DIM 1024
#define TK    32               // top_k (fixed by reference)
#define TM    32               // tokens per block (two WMMA M tiles, B-reuse)
#define NWG   4                // GEMM waves
#define NTHR  ((NWG + 1) * 32) // + 1 dedicated top-k wave = 160 threads
#define CT    (NWG * 16)       // 64 neuron columns per outer iteration
#define CTP   (CT + 1)         // padded stride -> conflict-free per-token rows
#define TKP   (TK + 1)

// Dynamic LDS layout (~93 KB):
//   xs   : TM*D_DIM f16          (64 KB)   WMMA A source
//   ltile: 2*TM*CTP f32 ping-pong(16.25KB) logits staging
//   topv : TM*TKP   f32          (4.1 KB)
//   topi : TM*TKP   i32          (4.1 KB)
//   wgt  : TM*TKP   f32          (4.1 KB)  gate*act
//   red  : TM       f32          (128 B)
#define SMEM_BYTES (TM*D_DIM*2 + 2*TM*CTP*4 + 3*TM*TKP*4 + TM*4)

// Fused: router GEMM (WMMA f16, 4 producer waves) overlapped with a dedicated
// top-k consumer wave via ping-pong LDS buffer; then softmax gate + combine.
template<bool KN16>
__global__ __launch_bounds__(NTHR) void kc_router_fused(
    const float* __restrict__ x, const float* __restrict__ kn,
    const _Float16* __restrict__ kn16,
    float* __restrict__ out, float* __restrict__ partial, int P)
{
    extern __shared__ char smem_raw[];
    _Float16* xs    = (_Float16*)smem_raw;                       // [TM][D_DIM]
    float*    ltile = (float*)(smem_raw + TM * D_DIM * 2);       // [2][TM][CTP]
    float*    topv  = ltile + 2 * TM * CTP;                      // [TM][TKP]
    int*      topi  = (int*)(topv + TM * TKP);                   // [TM][TKP]
    float*    wgt   = (float*)(topi + TM * TKP);                 // [TM][TKP]
    float*    red   = wgt + TM * TKP;                            // [TM]

    const int tid  = threadIdx.x;
    const int lane = tid & 31;
    const int wave = tid >> 5;
    const int tok0 = blockIdx.x * TM;

    // ---- stage x tile to LDS as f16 (vectorized, coalesced) ----
    {
        const float* xb = x + (size_t)tok0 * D_DIM;
        v4h* xsv = (v4h*)xs;
        for (int i = tid; i < TM * D_DIM / 4; i += NTHR) {
            float4 v = ((const float4*)xb)[i];
            v4h h = { (_Float16)v.x, (_Float16)v.y, (_Float16)v.z, (_Float16)v.w };
            xsv[i] = h;
        }
    }
    if (wave == NWG) {   // top-k wave: lane == token
        for (int n = 0; n < TK; ++n) { topv[lane*TKP + n] = -1e30f; topi[lane*TKP + n] = 0; }
    }
    __syncthreads();

    float vmin = -1e30f; int minpos = 0;   // per-lane top-k state (top-k wave)

    const int nsub = lane & 15;            // A row / B column within 16-tile
    const int hi   = lane >> 4;            // half-wave select
    const _Float16* arow0 = xs + (size_t)nsub * D_DIM;            // M tile 0
    const _Float16* arow1 = xs + (size_t)(16 + nsub) * D_DIM;     // M tile 1
    const int aoff = hi * 16;              // A: lanes 16-31 hold K=+16 group

    const int niter = P / CT;
    for (int it = 0; it <= niter; ++it) {
        if (wave < NWG) {
            // ================= producer: WMMA tile -> ltile[it&1] =============
            if (it < niter) {
                const int cb   = it * CT;
                const int col0 = cb + wave * 16;
                const float*    bp32 = kn   + (size_t)(col0 + nsub) * D_DIM + hi * 16;
                const _Float16* bp16 = KN16 ? (kn16 + (size_t)(col0 + nsub) * D_DIM + hi * 16)
                                            : (const _Float16*)nullptr;
                v8f c0 = {}, c1 = {};
                #pragma unroll 4
                for (int kk = 0; kk < D_DIM; kk += 32) {
                    v16h b;
                    if (KN16) {
                        v8h blo = *(const v8h*)(bp16 + kk);
                        v8h bhi = *(const v8h*)(bp16 + kk + 8);
                        b = __builtin_shufflevector(blo, bhi,
                            0,1,2,3,4,5,6,7,8,9,10,11,12,13,14,15);
                    } else {
                        float4 b0 = *(const float4*)(bp32 + kk + 0);
                        float4 b1 = *(const float4*)(bp32 + kk + 4);
                        float4 b2 = *(const float4*)(bp32 + kk + 8);
                        float4 b3 = *(const float4*)(bp32 + kk + 12);
                        b = (v16h){ (_Float16)b0.x, (_Float16)b0.y, (_Float16)b0.z, (_Float16)b0.w,
                                    (_Float16)b1.x, (_Float16)b1.y, (_Float16)b1.z, (_Float16)b1.w,
                                    (_Float16)b2.x, (_Float16)b2.y, (_Float16)b2.z, (_Float16)b2.w,
                                    (_Float16)b3.x, (_Float16)b3.y, (_Float16)b3.z, (_Float16)b3.w };
                    }
                    v8h a0lo = *(const v8h*)(arow0 + kk + aoff);
                    v8h a0hi = *(const v8h*)(arow0 + kk + aoff + 8);
                    v16h a0  = __builtin_shufflevector(a0lo, a0hi,
                               0,1,2,3,4,5,6,7,8,9,10,11,12,13,14,15);
                    v8h a1lo = *(const v8h*)(arow1 + kk + aoff);
                    v8h a1hi = *(const v8h*)(arow1 + kk + aoff + 8);
                    v16h a1  = __builtin_shufflevector(a1lo, a1hi,
                               0,1,2,3,4,5,6,7,8,9,10,11,12,13,14,15);
                    // one B fragment feeds two WMMAs (halves kn L2 traffic)
                    c0 = __builtin_amdgcn_wmma_f32_16x16x32_f16(false, a0, false, b,
                                                                (short)0, c0, false, false);
                    c1 = __builtin_amdgcn_wmma_f32_16x16x32_f16(false, a1, false, b,
                                                                (short)0, c1, false, false);
                }
                // C/D layout: VGPR j -> (M=j,N=lane) lanes 0-15; (M=j+8,N=lane-16) 16-31
                float* lt = ltile + (size_t)(it & 1) * TM * CTP;
                const int mbase = hi * 8;
                const int cidx  = wave * 16 + nsub;
                #pragma unroll
                for (int j = 0; j < 8; ++j) {
                    lt[(mbase + j)      * CTP + cidx] = c0[j];
                    lt[(16 + mbase + j) * CTP + cidx] = c1[j];
                }
            }
        } else {
            // ============ consumer: top-k scan of previous tile ===============
            if (it > 0) {
                const int cbase = (it - 1) * CT;
                const float* lr = ltile + (size_t)((it - 1) & 1) * TM * CTP + lane * CTP;
                float* tv = topv + lane * TKP;
                int*   ti = topi + lane * TKP;
                for (int j = 0; j < CT; ++j) {
                    float l = lr[j];
                    if (l > vmin) {
                        tv[minpos] = l;
                        ti[minpos] = cbase + j;
                        float m = tv[0]; int mp = 0;
                        #pragma unroll
                        for (int n = 1; n < TK; ++n) {
                            float t2 = tv[n];
                            if (t2 < m) { m = t2; mp = n; }
                        }
                        vmin = m; minpos = mp;
                    }
                }
            }
        }
        __syncthreads();
    }

    // ---- softmax gate, combine weights (gate*act), aux partial ----
    if (wave == NWG) {
        float* tv = topv + lane * TKP;
        float* tw = wgt  + lane * TKP;
        float mx = -1e30f;
        for (int n = 0; n < TK; ++n) mx = fmaxf(mx, tv[n]);
        float s = 0.f, lsum = 0.f;
        for (int n = 0; n < TK; ++n) {
            float v = tv[n];
            lsum += v;
            float e = expf(v - mx);
            tw[n] = e; s += e;
        }
        float inv = 1.f / s;
        for (int n = 0; n < TK; ++n) tw[n] = tw[n] * inv * tv[n];
        red[lane] = lsum;
    }
    __syncthreads();
    if (tid == 0) {
        float s = 0.f;
        for (int t = 0; t < TM; ++t) s += red[t];
        partial[blockIdx.x] = s;
    }

    // ---- combine: out[t][:] = sum_n wgt[t][n] * kn[topi[t][n]][:] ----
    // 8-float chunks; gathered kn rows are L2-resident (f32, exact)
    for (int t = 0; t < TM; ++t) {
        for (int i = tid; i < D_DIM / 8; i += NTHR) {
            float ax=0,ay=0,az=0,aw=0, bx=0,by=0,bz=0,bw=0;
            #pragma unroll 4
            for (int n = 0; n < TK; ++n) {
                float w = wgt[t*TKP + n];
                const float4* r = (const float4*)(kn + (size_t)topi[t*TKP + n] * D_DIM) + i * 2;
                float4 r0 = r[0], r1 = r[1];
                ax += w*r0.x; ay += w*r0.y; az += w*r0.z; aw += w*r0.w;
                bx += w*r1.x; by += w*r1.y; bz += w*r1.z; bw += w*r1.w;
            }
            float4* o = (float4*)(out + (size_t)(tok0 + t) * D_DIM) + i * 2;
            o[0] = make_float4(ax, ay, az, aw);
            o[1] = make_float4(bx, by, bz, bw);
        }
    }
}

// One-shot fp32 -> fp16 conversion of the neuron pool into workspace.
__global__ __launch_bounds__(256) void kc_cvt_kn(
    const float* __restrict__ kn, _Float16* __restrict__ kn16, int n4)
{
    int i = blockIdx.x * 256 + threadIdx.x;
    if (i < n4) {
        float4 v = ((const float4*)kn)[i];
        v4h h = { (_Float16)v.x, (_Float16)v.y, (_Float16)v.z, (_Float16)v.w };
        ((v4h*)kn16)[i] = h;
    }
}

// Deterministic fixed-order reduction of per-block aux partials.
__global__ __launch_bounds__(256) void kc_aux_reduce(
    const float* __restrict__ partial, int n, float* __restrict__ auxp, float scale)
{
    __shared__ float s[256];
    int tid = threadIdx.x;
    float v = 0.f;
    for (int i = tid; i < n; i += 256) v += partial[i];
    s[tid] = v;
    __syncthreads();
    for (int st = 128; st > 0; st >>= 1) {
        if (tid < st) s[tid] += s[tid + st];
        __syncthreads();
    }
    if (tid == 0) *auxp = s[0] * scale;
}

extern "C" void kernel_launch(void* const* d_in, const int* in_sizes, int n_in,
                              void* d_out, int out_size, void* d_ws, size_t ws_size,
                              hipStream_t stream) {
    const float* x  = (const float*)d_in[0];   // [B,S,D] f32
    const float* kn = (const float*)d_in[1];   // [P,D]   f32
    // d_in[2] attention_mask: all-ones, unused by reference math
    // d_in[3] top_k: fixed 32 (TK)
    float* out = (float*)d_out;                // [B*S*D] then 1 aux scalar

    const int BS = in_sizes[0] / D_DIM;        // 4096 tokens
    const int P  = in_sizes[1] / D_DIM;        // 16384 neurons
    const int nblocks = BS / TM;               // 128

    const size_t kn16_bytes = (size_t)P * D_DIM * sizeof(_Float16);      // 32 MB
    const size_t part_off   = (kn16_bytes + 255) & ~(size_t)255;
    const bool   use16      = ws_size >= part_off + (size_t)nblocks * sizeof(float);

    if (use16) {
        _Float16* kn16    = (_Float16*)d_ws;
        float*    partial = (float*)((char*)d_ws + part_off);
        const int n4 = P * D_DIM / 4;
        kc_cvt_kn<<<(n4 + 255) / 256, 256, 0, stream>>>(kn, kn16, n4);
        kc_router_fused<true><<<nblocks, NTHR, SMEM_BYTES, stream>>>(
            x, kn, kn16, out, partial, P);
        kc_aux_reduce<<<1, 256, 0, stream>>>(partial, nblocks,
                                             out + (size_t)BS * D_DIM,
                                             1.0f / (float)(BS * TK));
    } else {
        float* partial = (float*)d_ws;
        kc_router_fused<false><<<nblocks, NTHR, SMEM_BYTES, stream>>>(
            x, kn, nullptr, out, partial, P);
        kc_aux_reduce<<<1, 256, 0, stream>>>(partial, nblocks,
                                             out + (size_t)BS * D_DIM,
                                             1.0f / (float)(BS * TK));
    }
}
